// AttentionBlock_21199958573383
// MI455X (gfx1250) — matrix operand order repaired and checked by reading, not compiled, not run on hardware
//
#include <hip/hip_runtime.h>
#include <hip/hip_bf16.h>
#include <math.h>

// ---------------------------------------------------------------------------
// Types
// ---------------------------------------------------------------------------
typedef __attribute__((ext_vector_type(16))) __bf16 v16bf;
typedef __attribute__((ext_vector_type(8)))  float  v8f;
typedef __attribute__((ext_vector_type(8)))  __bf16 bf16x8;
typedef __attribute__((ext_vector_type(4)))  __bf16 bf16x4;

#define B_   2
#define S_   2048
#define D_   1024
#define H_   16
#define HD_  64
#define MLP_ 4096
#define ROWS_ (B_ * S_)          // 4096
#define BSD_  (B_ * S_ * D_)     // 4194304

__device__ inline v8f vzero8() {
  v8f r;
#pragma unroll
  for (int i = 0; i < 8; ++i) r[i] = 0.0f;
  return r;
}

// ---------------------------------------------------------------------------
// CDNA5 async global->LDS copy (16B per lane), ASYNCcnt-tracked.
// ---------------------------------------------------------------------------
__device__ inline void async_copy_b128(const void* gptr, const void* lptr) {
  unsigned lds = (unsigned)(unsigned long long)(uintptr_t)lptr;  // low 32b = LDS addr
  asm volatile("global_load_async_to_lds_b128 %0, %1, off"
               :: "v"(lds), "v"(gptr) : "memory");
}
#define ASYNC_WAIT_0() asm volatile("s_wait_asynccnt 0x0" ::: "memory")
#define ASYNC_WAIT_4() asm volatile("s_wait_asynccnt 0x4" ::: "memory")
#define ASYNC_WAIT_8() asm volatile("s_wait_asynccnt 0x8" ::: "memory")

// A-fragment (16x32 bf16): lane holds row (lane&15); K chunks [kb,kb+8) and
// [kb+16,kb+24) with kb = 8*(lane>>4).  Two 16B LDS loads.
__device__ inline v16bf load_frag_A(const __bf16* base, int ld) {
  int lane = threadIdx.x & 31;
  int row  = lane & 15;
  int kb   = (lane >> 4) * 8;
  const __bf16* p = base + row * ld + kb;
  union { v16bf v; bf16x8 h[2]; } u;
  u.h[0] = *reinterpret_cast<const bf16x8*>(p);
  u.h[1] = *reinterpret_cast<const bf16x8*>(p + 16);
  return u.v;
}

// B-fragment (32x16 bf16) from column-major LDS ([n][k], stride ld): lane holds
// col (lane&15); K chunks [kb,kb+8) and [kb+16,kb+24) with kb = 8*(lane>>4).  Two 16B LDS loads.
__device__ inline v16bf load_frag_B(const __bf16* base, int ld) {
  int lane = threadIdx.x & 31;
  int col  = lane & 15;
  int kb   = (lane >> 4) * 8;
  const __bf16* p = base + col * ld + kb;
  union { v16bf v; bf16x8 h[2]; } u;
  u.h[0] = *reinterpret_cast<const bf16x8*>(p);
  u.h[1] = *reinterpret_cast<const bf16x8*>(p + 16);
  return u.v;
}

// ---------------------------------------------------------------------------
// Weight transpose + fp32->bf16:  out[n*K + k] = (bf16) in[k*N + n]
// 32x32 tiles, 256 threads (32x8).
// ---------------------------------------------------------------------------
__global__ __launch_bounds__(256) void transpose_cvt_kernel(
    const float* __restrict__ in, __bf16* __restrict__ out, int K, int N) {
  __shared__ float tile[32][33];
  int n0 = blockIdx.x * 32, k0 = blockIdx.y * 32;
  int tx = threadIdx.x & 31, ty = threadIdx.x >> 5;
#pragma unroll
  for (int i = 0; i < 4; ++i) {
    int r = ty + i * 8;
    tile[r][tx] = in[(size_t)(k0 + r) * N + n0 + tx];
  }
  __syncthreads();
#pragma unroll
  for (int i = 0; i < 4; ++i) {
    int r = ty + i * 8;
    out[(size_t)(n0 + r) * K + k0 + tx] = (__bf16)tile[tx][r];
  }
}

// ---------------------------------------------------------------------------
// LayerNorm over D=1024; optional second input (residual add); fp32 and/or
// bf16 outputs.  One 256-thread block per row, 4 elems/thread.
// ---------------------------------------------------------------------------
__global__ __launch_bounds__(256) void ln_kernel(
    const float* __restrict__ in, const float* __restrict__ in2,
    const float* __restrict__ gam, const float* __restrict__ bet,
    float* __restrict__ outF, __bf16* __restrict__ outB) {
  __shared__ float red[16];
  int row = blockIdx.x;
  int tid = threadIdx.x;
  const float4 xv = reinterpret_cast<const float4*>(in + (size_t)row * D_)[tid];
  float x0 = xv.x, x1 = xv.y, x2 = xv.z, x3 = xv.w;
  if (in2) {
    const float4 yv = reinterpret_cast<const float4*>(in2 + (size_t)row * D_)[tid];
    x0 += yv.x; x1 += yv.y; x2 += yv.z; x3 += yv.w;
  }
  float s = x0 + x1 + x2 + x3;
#pragma unroll
  for (int o = 1; o < 32; o <<= 1) s += __shfl_xor(s, o, 32);
  if ((tid & 31) == 0) red[tid >> 5] = s;
  __syncthreads();
  float tot = 0.0f;
#pragma unroll
  for (int i = 0; i < 8; ++i) tot += red[i];
  float mu = tot * (1.0f / D_);
  float d0 = x0 - mu, d1 = x1 - mu, d2 = x2 - mu, d3 = x3 - mu;
  float vs = d0 * d0 + d1 * d1 + d2 * d2 + d3 * d3;
#pragma unroll
  for (int o = 1; o < 32; o <<= 1) vs += __shfl_xor(vs, o, 32);
  if ((tid & 31) == 0) red[8 + (tid >> 5)] = vs;
  __syncthreads();
  float vtot = 0.0f;
#pragma unroll
  for (int i = 0; i < 8; ++i) vtot += red[8 + i];
  float rstd = rsqrtf(vtot * (1.0f / D_) + 1e-5f);

  int c = tid * 4;
  float y0 = d0 * rstd * gam[c + 0] + bet[c + 0];
  float y1 = d1 * rstd * gam[c + 1] + bet[c + 1];
  float y2 = d2 * rstd * gam[c + 2] + bet[c + 2];
  float y3 = d3 * rstd * gam[c + 3] + bet[c + 3];
  if (outF) {
    float4 o4 = make_float4(y0, y1, y2, y3);
    reinterpret_cast<float4*>(outF + (size_t)row * D_)[tid] = o4;
  }
  if (outB) {
    bf16x4 ob;
    ob[0] = (__bf16)y0; ob[1] = (__bf16)y1; ob[2] = (__bf16)y2; ob[3] = (__bf16)y3;
    reinterpret_cast<bf16x4*>(outB + (size_t)row * D_)[tid] = ob;
  }
}

// ---------------------------------------------------------------------------
// Tiled bf16 GEMM: C[M,N] = A[M,K] * Wt[N,K]^T + bias.
// Block tile 128x128x32, 256 threads (8 waves, each 64x32 = 8 WMMA accums).
// Double-buffered async global->LDS staging (pure row-major copies).
// Epilogue: optional exact GELU, fp32 residual add, fp32 / bf16 / per-head-
// transposed-bf16 outputs.
// ---------------------------------------------------------------------------
#define GBM 128
#define GBN 128
#define GBK 32
#define GLD 40   // 80B row stride (16B multiple)

__global__ __launch_bounds__(256) void gemm_bf16_kernel(
    const __bf16* __restrict__ A, const __bf16* __restrict__ Wt,
    const float* __restrict__ bias,
    float* __restrict__ outF, __bf16* __restrict__ outB,
    __bf16* __restrict__ outVT,            // [b][h][hd][S] scatter (V only)
    const float* __restrict__ res,
    int M, int Ncols, int K, int gelu) {
  __shared__ __attribute__((aligned(16))) __bf16 sA[2][GBM * GLD];
  __shared__ __attribute__((aligned(16))) __bf16 sB[2][GBN * GLD];
  int tid  = threadIdx.x;
  int lane = tid & 31;
  int wave = tid >> 5;
  int n0 = blockIdx.x * GBN;
  int m0 = blockIdx.y * GBM;
  int wm = (wave >> 2) * 64;   // 2 wave-rows
  int wn = (wave & 3) * 32;    // 4 wave-cols

  v8f acc[4][2];
#pragma unroll
  for (int mi = 0; mi < 4; ++mi)
#pragma unroll
    for (int ni = 0; ni < 2; ++ni) acc[mi][ni] = vzero8();

  // async stage of one K-tile (A: 128x32, B: 128x32) into buffer `bi`
  auto stage = [&](int kt, int bi) {
#pragma unroll
    for (int it = 0; it < 2; ++it) {
      int c = tid + it * 256;            // chunk id 0..511 (16B chunks)
      int r = c >> 2, cc = (c & 3) * 8;  // row, elem offset
      async_copy_b128(A + (size_t)(m0 + r) * K + kt + cc, &sA[bi][r * GLD + cc]);
    }
#pragma unroll
    for (int it = 0; it < 2; ++it) {
      int c = tid + it * 256;
      int r = c >> 2, cc = (c & 3) * 8;
      async_copy_b128(Wt + (size_t)(n0 + r) * K + kt + cc, &sB[bi][r * GLD + cc]);
    }
  };

  stage(0, 0);
  int nk = K / GBK;
  for (int i = 0; i < nk; ++i) {
    int cur = i & 1;
    if (i + 1 < nk) { stage((i + 1) * GBK, cur ^ 1); ASYNC_WAIT_4(); }
    else           { ASYNC_WAIT_0(); }
    __syncthreads();

    v16bf af[4], bfg[2];
#pragma unroll
    for (int mi = 0; mi < 4; ++mi)
      af[mi] = load_frag_A(&sA[cur][(wm + mi * 16) * GLD], GLD);
#pragma unroll
    for (int ni = 0; ni < 2; ++ni)
      bfg[ni] = load_frag_B(&sB[cur][(wn + ni * 16) * GLD], GLD);
#pragma unroll
    for (int mi = 0; mi < 4; ++mi)
#pragma unroll
      for (int ni = 0; ni < 2; ++ni)
        acc[mi][ni] = __builtin_amdgcn_wmma_f32_16x16x32_bf16(
            false, af[mi], false, bfg[ni], (short)0, acc[mi][ni], false, false);
    __syncthreads();
  }

  // Epilogue.  C layout: VGPR r, lane -> (M = r + 8*(lane>>4), N = lane&15)
  int lh = lane >> 4, lc = lane & 15;
#pragma unroll
  for (int mi = 0; mi < 4; ++mi)
#pragma unroll
    for (int ni = 0; ni < 2; ++ni) {
      int ncol = n0 + wn + ni * 16 + lc;
      float bval = bias[ncol];
#pragma unroll
      for (int r = 0; r < 8; ++r) {
        int mrow = m0 + wm + mi * 16 + r + 8 * lh;
        float val = acc[mi][ni][r] + bval;
        if (gelu) val = 0.5f * val * (1.0f + erff(val * 0.70710678f));
        size_t o = (size_t)mrow * Ncols + ncol;
        if (res)  val += res[o];
        if (outF) outF[o] = val;
        if (outB) outB[o] = (__bf16)val;
        if (outVT) {
          // scatter into per-head transposed layout [b][h][hd][S]
          int b  = mrow >> 11;            // /2048
          int sl = mrow & 2047;
          int h  = sl >> 7;               // view head
          int sp = (sl & 127) * 16 + (ncol >> 6);
          int hd = ncol & 63;
          outVT[(((size_t)(b * H_ + h) * HD_) + hd) * S_ + sp] = (__bf16)val;
        }
      }
    }
}

// ---------------------------------------------------------------------------
// Flash attention over the contiguous per-head [S,64] view.  V comes in
// pre-transposed [b][h][hd][S], so ALL LDS staging is a straight async copy.
// grid = (S/64 query tiles, B*H); block = 128 (4 waves x 16 query rows).
// Double-buffered K/VT tiles.
// ---------------------------------------------------------------------------
__global__ __launch_bounds__(128) void attn_kernel(
    const __bf16* __restrict__ q, const __bf16* __restrict__ k,
    const __bf16* __restrict__ vT,
    const float* __restrict__ dmask, const float* __restrict__ cmask,
    float* __restrict__ out) {
  __shared__ __attribute__((aligned(16))) __bf16 sQ[64 * 72];
  __shared__ __attribute__((aligned(16))) __bf16 sK[2][64 * 72];
  __shared__ __attribute__((aligned(16))) __bf16 sVT[2][64 * 72];
  __shared__ __attribute__((aligned(16))) __bf16 sP[4][16 * 72];

  int qt = blockIdx.x;
  int bh = blockIdx.y;
  int b = bh >> 4;
  const size_t headOff = (size_t)bh * (S_ * HD_);
  const __bf16* Q  = q + headOff;
  const __bf16* K  = k + headOff;
  const __bf16* VT = vT + headOff;   // [64 hd][2048 t]

  int tid = threadIdx.x, lane = tid & 31, wave = tid >> 5;
  int lh = lane >> 4, lc = lane & 15;

  // stage K/VT tile (keys t0..t0+63) into buffer bi:  8 async per wave
  auto stageKV = [&](int t0, int bi) {
#pragma unroll
    for (int it = 0; it < 4; ++it) {
      int c = tid + it * 128;            // 0..511
      int r = c >> 3, cc = (c & 7) * 8;  // key row, d offset
      async_copy_b128(K + (size_t)(t0 + r) * HD_ + cc, &sK[bi][r * 72 + cc]);
    }
#pragma unroll
    for (int it = 0; it < 4; ++it) {
      int c = tid + it * 128;
      int r = c >> 3, cc = (c & 7) * 8;  // hd row, t offset
      async_copy_b128(VT + (size_t)r * S_ + t0 + cc, &sVT[bi][r * 72 + cc]);
    }
  };

  // stage Q tile (4 async per wave), then K/V tile 0
#pragma unroll
  for (int it = 0; it < 4; ++it) {
    int c = tid + it * 128;
    int r = c >> 3, cc = (c & 7) * 8;
    async_copy_b128(Q + (size_t)(qt * 64 + r) * HD_ + cc, &sQ[r * 72 + cc]);
  }
  stageKV(0, 0);
  ASYNC_WAIT_8();   // Q (first 4) complete; K/V tile0 may still be in flight
  __syncthreads();

  const __bf16* qbase = &sQ[wave * 16 * 72];
  v16bf qf[2];
  qf[0] = load_frag_A(qbase, 72);        // d 0..31
  qf[1] = load_frag_A(qbase + 32, 72);   // d 32..63

  float mrun[8], lrun[8];
  v8f o[4];
#pragma unroll
  for (int r = 0; r < 8; ++r) { mrun[r] = -1e30f; lrun[r] = 0.0f; }
#pragma unroll
  for (int dt = 0; dt < 4; ++dt) o[dt] = vzero8();

  int srow_base = qt * 64 + wave * 16;
  const float scale = 0.125f;            // 1/sqrt(64)
  const float LOG2E = 1.4426950408889634f;

  for (int i = 0; i < S_ / 64; ++i) {
    int t0 = i * 64;
    int cur = i & 1;
    if (i + 1 < S_ / 64) { stageKV(t0 + 64, cur ^ 1); ASYNC_WAIT_8(); }
    else                 { ASYNC_WAIT_0(); }
    __syncthreads();

    // Scores: 16 rows x 64 keys per wave = four 16x16 C frags
    v8f sc[4];
#pragma unroll
    for (int nt = 0; nt < 4; ++nt) {
      v8f c = vzero8();
      v16bf b0 = load_frag_B(&sK[cur][nt * 16 * 72], 72);        // d 0..31
      v16bf b1 = load_frag_B(&sK[cur][nt * 16 * 72] + 32, 72);   // d 32..63
      c = __builtin_amdgcn_wmma_f32_16x16x32_bf16(false, qf[0], false, b0, (short)0, c, false, false);
      c = __builtin_amdgcn_wmma_f32_16x16x32_bf16(false, qf[1], false, b1, (short)0, c, false, false);
      sc[nt] = c;
    }
    // scale + masks
#pragma unroll
    for (int nt = 0; nt < 4; ++nt) {
      int trow = t0 + nt * 16 + lc;
#pragma unroll
      for (int r = 0; r < 8; ++r) {
        int srow = srow_base + r + 8 * lh;
        size_t moff = (size_t)b * S_ * S_ + (size_t)srow * S_ + trow;
        sc[nt][r] = sc[nt][r] * scale + dmask[moff] + cmask[moff];
      }
    }
    // row max (rows live across one 16-lane half)
    float tm[8];
#pragma unroll
    for (int r = 0; r < 8; ++r) {
      tm[r] = fmaxf(fmaxf(sc[0][r], sc[1][r]), fmaxf(sc[2][r], sc[3][r]));
#pragma unroll
      for (int off = 1; off < 16; off <<= 1)
        tm[r] = fmaxf(tm[r], __shfl_xor(tm[r], off, 32));
    }
    float alpha[8];
#pragma unroll
    for (int r = 0; r < 8; ++r) {
      float mnew = fmaxf(mrun[r], tm[r]);
      alpha[r] = exp2f((mrun[r] - mnew) * LOG2E);
      mrun[r] = mnew;
    }
    // p = exp(s - m), row sums
    float rs[8];
#pragma unroll
    for (int r = 0; r < 8; ++r) rs[r] = 0.0f;
#pragma unroll
    for (int nt = 0; nt < 4; ++nt)
#pragma unroll
      for (int r = 0; r < 8; ++r) {
        float p = exp2f((sc[nt][r] - mrun[r]) * LOG2E);
        sc[nt][r] = p;
        rs[r] += p;
      }
#pragma unroll
    for (int r = 0; r < 8; ++r) {
#pragma unroll
      for (int off = 1; off < 16; off <<= 1) rs[r] += __shfl_xor(rs[r], off, 32);
      lrun[r] = lrun[r] * alpha[r] + rs[r];
    }
#pragma unroll
    for (int dt = 0; dt < 4; ++dt)
#pragma unroll
      for (int r = 0; r < 8; ++r) o[dt][r] *= alpha[r];

    // P (C layout) -> per-wave LDS -> A layout
    __bf16* pw = sP[wave];
#pragma unroll
    for (int nt = 0; nt < 4; ++nt)
#pragma unroll
      for (int r = 0; r < 8; ++r)
        pw[(r + 8 * lh) * 72 + nt * 16 + lc] = (__bf16)sc[nt][r];
    asm volatile("s_wait_dscnt 0x0" ::: "memory");
    v16bf pa0 = load_frag_A(pw, 72);        // t 0..31
    v16bf pa1 = load_frag_A(pw + 32, 72);   // t 32..63
#pragma unroll
    for (int dt = 0; dt < 4; ++dt) {
      v16bf vb0 = load_frag_B(&sVT[cur][dt * 16 * 72], 72);        // t 0..31
      v16bf vb1 = load_frag_B(&sVT[cur][dt * 16 * 72] + 32, 72);   // t 32..63
      o[dt] = __builtin_amdgcn_wmma_f32_16x16x32_bf16(false, pa0, false, vb0, (short)0, o[dt], false, false);
      o[dt] = __builtin_amdgcn_wmma_f32_16x16x32_bf16(false, pa1, false, vb1, (short)0, o[dt], false, false);
    }
    __syncthreads();
  }

  // Normalize and write fp32 output in the contiguous head layout
  float* O = out + headOff;
#pragma unroll
  for (int r = 0; r < 8; ++r) {
    float inv = 1.0f / lrun[r];
    int srow = srow_base + r + 8 * lh;
#pragma unroll
    for (int dt = 0; dt < 4; ++dt) {
      int d = dt * 16 + lc;
      O[(size_t)srow * HD_ + d] = o[dt][r] * inv;
    }
  }
}

// ---------------------------------------------------------------------------
// Launch
// ---------------------------------------------------------------------------
extern "C" void kernel_launch(void* const* d_in, const int* in_sizes, int n_in,
                              void* d_out, int out_size, void* d_ws, size_t ws_size,
                              hipStream_t stream) {
  (void)in_sizes; (void)n_in; (void)out_size; (void)ws_size;
  const float* x     = (const float*)d_in[0];
  const float* dmask = (const float*)d_in[1];
  const float* cmask = (const float*)d_in[2];
  const float* wq  = (const float*)d_in[3];
  const float* bq  = (const float*)d_in[4];
  const float* wk  = (const float*)d_in[5];
  const float* bk  = (const float*)d_in[6];
  const float* wv  = (const float*)d_in[7];
  const float* bv  = (const float*)d_in[8];
  const float* l1g = (const float*)d_in[9];
  const float* l1b = (const float*)d_in[10];
  const float* l2g = (const float*)d_in[11];
  const float* l2b = (const float*)d_in[12];
  const float* w1  = (const float*)d_in[13];
  const float* b1  = (const float*)d_in[14];
  const float* w2  = (const float*)d_in[15];
  const float* b2  = (const float*)d_in[16];
  float* out = (float*)d_out;

  char* ws = (char*)d_ws;
  size_t off = 0;
  auto alloc = [&](size_t bytes) -> void* {
    void* p = ws + off;
    off += (bytes + 255) & ~(size_t)255;
    return p;
  };
  __bf16* h1   = (__bf16*)alloc((size_t)BSD_ * 2);
  __bf16* wqT  = (__bf16*)alloc((size_t)D_ * D_ * 2);
  __bf16* wkT  = (__bf16*)alloc((size_t)D_ * D_ * 2);
  __bf16* wvT  = (__bf16*)alloc((size_t)D_ * D_ * 2);
  __bf16* w1T  = (__bf16*)alloc((size_t)D_ * MLP_ * 2);
  __bf16* w2T  = (__bf16*)alloc((size_t)MLP_ * D_ * 2);
  __bf16* qB   = (__bf16*)alloc((size_t)BSD_ * 2);
  __bf16* kB   = (__bf16*)alloc((size_t)BSD_ * 2);
  __bf16* vTB  = (__bf16*)alloc((size_t)BSD_ * 2);
  float*  attn = (float*)alloc((size_t)BSD_ * 4);
  float*  l2F  = (float*)alloc((size_t)BSD_ * 4);
  __bf16* l2B  = (__bf16*)alloc((size_t)BSD_ * 2);
  __bf16* m1   = (__bf16*)alloc((size_t)ROWS_ * MLP_ * 2);

  // 1) weight transposes (+bf16): wT[n][k]
  transpose_cvt_kernel<<<dim3(D_ / 32, D_ / 32), 256, 0, stream>>>(wq, wqT, D_, D_);
  transpose_cvt_kernel<<<dim3(D_ / 32, D_ / 32), 256, 0, stream>>>(wk, wkT, D_, D_);
  transpose_cvt_kernel<<<dim3(D_ / 32, D_ / 32), 256, 0, stream>>>(wv, wvT, D_, D_);
  transpose_cvt_kernel<<<dim3(MLP_ / 32, D_ / 32), 256, 0, stream>>>(w1, w1T, D_, MLP_);
  transpose_cvt_kernel<<<dim3(D_ / 32, MLP_ / 32), 256, 0, stream>>>(w2, w2T, MLP_, D_);
  // 2) LN1
  ln_kernel<<<ROWS_, 256, 0, stream>>>(x, nullptr, l1g, l1b, nullptr, h1);
  // 3) QKV projections (V scattered into per-head transposed layout)
  dim3 gqkv(D_ / GBN, ROWS_ / GBM);
  gemm_bf16_kernel<<<gqkv, 256, 0, stream>>>(h1, wqT, bq, nullptr, qB, nullptr, nullptr, ROWS_, D_, D_, 0);
  gemm_bf16_kernel<<<gqkv, 256, 0, stream>>>(h1, wkT, bk, nullptr, kB, nullptr, nullptr, ROWS_, D_, D_, 0);
  gemm_bf16_kernel<<<gqkv, 256, 0, stream>>>(h1, wvT, bv, nullptr, nullptr, vTB, nullptr, ROWS_, D_, D_, 0);
  // 4) attention
  attn_kernel<<<dim3(S_ / 64, B_ * H_), 128, 0, stream>>>(qB, kB, vTB, dmask, cmask, attn);
  // 5) residual + LN2 (fp32 kept for final residual, bf16 for GEMM)
  ln_kernel<<<ROWS_, 256, 0, stream>>>(attn, x, l2g, l2b, l2F, l2B);
  // 6) MLP1 + exact GELU (bf16 out)
  gemm_bf16_kernel<<<dim3(MLP_ / GBN, ROWS_ / GBM), 256, 0, stream>>>(
      l2B, w1T, b1, nullptr, m1, nullptr, nullptr, ROWS_, MLP_, D_, 1);
  // 7) MLP2 + bias + residual -> final fp32 output
  gemm_bf16_kernel<<<dim3(D_ / GBN, ROWS_ / GBM), 256, 0, stream>>>(
      m1, w2T, b2, out, nullptr, nullptr, l2F, ROWS_, D_, MLP_, 0);
}
